// PerformerAttention_53781580480929
// MI455X (gfx1250) — matrix operand order repaired and checked by reading, not compiled, not run on hardware
//
#include <hip/hip_runtime.h>
#include <hip/hip_bf16.h>
#include <stddef.h>

// ---------------------------------------------------------------------------
// Performer (FAVOR+) attention for MI455X (gfx1250), wave32 + WMMA bf16.
//
//   1) convert x -> bf16; W{q,k,v} -> concatenated transposed bf16 [3072,1024]
//   2) QKV GEMM: double-buffered GLOBAL_LOAD_ASYNC_TO_LDS_B128 staging
//      (ASYNCcnt + barrier), v_wmma_f32_16x16x32_bf16 from LDS fragments.
//      Writes q,k as [bh,n,64] bf16 and v transposed [bh,64,n] bf16.
//   3) row norms 0.5*||q||^2, 0.5*||k||^2
//   4) kv kernel (grid bh x 8 n-splits): fused phi(k) (WMMA + exp -> LDS
//      pk^T tile) feeding kv[m,d] accumulation; fp32 partials + s partials.
//   5) kv reduce: fold 8 partials -> kv^T [bh,64,256] bf16, s[bh,256] fp32.
//   6) out kernel: fused phi(q) (WMMA + exp -> wave-private LDS pq tile),
//      z = pq . s from LDS, out GEMM pq x kv, divide, store fp32 [b,n,1024].
// ---------------------------------------------------------------------------

typedef __bf16 bf16_t;
typedef __attribute__((ext_vector_type(16))) __bf16 v16bf;
typedef __attribute__((ext_vector_type(8)))  __bf16 v8bf;
typedef __attribute__((ext_vector_type(8)))  float  v8f;

// Async-to-LDS builtin operand types: per hipcc diagnostic the builtin wants
// `int __vector_size__(16)` pointers in AS1 (global) / AS3 (LDS).
typedef int v4i_raw __attribute__((vector_size(16)));
typedef __attribute__((address_space(1))) v4i_raw as1_v4i;
typedef __attribute__((address_space(3))) v4i_raw as3_v4i;

#if defined(__has_builtin)
#if __has_builtin(__builtin_amdgcn_global_load_async_to_lds_b128) && \
    __has_builtin(__builtin_amdgcn_s_wait_asynccnt)
#define USE_ASYNC_LDS 1
#endif
#endif

#define BATCH 4
#define NSEQ  4096
#define DIM   1024
#define HEADS 16
#define HD    64
#define MFEAT 256
#define BH    (BATCH * HEADS)          // 64
#define ROWS  (BATCH * NSEQ)           // 16384
#define NSPLIT 8                       // kv n-splits
#define RSQRT_M 0.0625f                // 1/sqrt(256)

static __device__ __forceinline__ v8f vzero8() {
  v8f z = {0.f, 0.f, 0.f, 0.f, 0.f, 0.f, 0.f, 0.f};
  return z;
}

// A fragment (16x32 bf16, M x K). ISA layout: lane L (m=L%16, g=L/16) holds
// K chunks [k0+8g, k0+8g+8) and [k0+16+8g, k0+16+8g+8) of row m.
static __device__ __forceinline__ v16bf load_a_frag(const bf16_t* __restrict__ base,
                                                    int lda, int row0, int k0) {
  const int lane = threadIdx.x & 31;
  const int m = lane & 15, g = lane >> 4;
  const bf16_t* p = base + (size_t)(row0 + m) * lda + (k0 + 8 * g);
  v8bf lo = *(const v8bf*)(p);
  v8bf hi = *(const v8bf*)(p + 16);
  return __builtin_shufflevector(lo, hi, 0, 1, 2, 3, 4, 5, 6, 7,
                                 8, 9, 10, 11, 12, 13, 14, 15);
}

// B fragment (32x16 bf16, K x N) sourced from B^T stored row-major [N, K]:
// lane L (n=L%16, g=L/16) holds K chunks [k0+8g, k0+8g+8) and [k0+16+8g, k0+16+8g+8).
static __device__ __forceinline__ v16bf load_b_frag(const bf16_t* __restrict__ baseT,
                                                    int ldb, int col0, int k0) {
  const int lane = threadIdx.x & 31;
  const int n = lane & 15, g = lane >> 4;
  const bf16_t* p = baseT + (size_t)(col0 + n) * ldb + (k0 + 8 * g);
  v8bf lo = *(const v8bf*)(p);
  v8bf hi = *(const v8bf*)(p + 16);
  return __builtin_shufflevector(lo, hi, 0, 1, 2, 3, 4, 5, 6, 7,
                                 8, 9, 10, 11, 12, 13, 14, 15);
}

static __device__ __forceinline__ v8f wmma_bf16(v16bf a, v16bf b, v8f c) {
  return __builtin_amdgcn_wmma_f32_16x16x32_bf16(false, a, false, b,
                                                 (short)0, c, false, false);
}

// ------------------------------ converters --------------------------------

__global__ void perf_cvt_x(const float* __restrict__ x, bf16_t* __restrict__ xb, int n) {
  int i = blockIdx.x * blockDim.x + threadIdx.x;
  if (i < n) xb[i] = (bf16_t)x[i];
}

// wcat[(w*1024 + nn)*1024 + kk] = W_w[kk*1024 + nn]  (bf16, transposed, concat q|k|v)
__global__ void perf_cvt_wt(const float* __restrict__ Wq, const float* __restrict__ Wk,
                            const float* __restrict__ Wv, bf16_t* __restrict__ wcat) {
  int i = blockIdx.x * blockDim.x + threadIdx.x;
  if (i >= 3 * DIM * DIM) return;
  int w  = i >> 20;
  int r  = i & (DIM * DIM - 1);
  int nn = r >> 10;
  int kk = r & (DIM - 1);
  const float* W = (w == 0) ? Wq : (w == 1) ? Wk : Wv;
  wcat[i] = (bf16_t)W[(size_t)kk * DIM + nn];
}

__global__ void perf_cvt_omega(const float* __restrict__ om, bf16_t* __restrict__ omb, int n) {
  int i = blockIdx.x * blockDim.x + threadIdx.x;
  if (i < n) omb[i] = (bf16_t)om[i];
}

// 0.5 * ||row||^2 for [BH*NSEQ, 64] bf16 rows
__global__ void perf_norms(const bf16_t* __restrict__ t, float* __restrict__ norm, int nrows) {
  int row = blockIdx.x * blockDim.x + threadIdx.x;
  if (row >= nrows) return;
  const bf16_t* p = t + (size_t)row * HD;
  float s = 0.f;
#pragma unroll
  for (int i = 0; i < HD; ++i) { float f = (float)p[i]; s += f * f; }
  norm[row] = 0.5f * s;
}

// ------------------------------ QKV GEMM ----------------------------------
// C[16384, 3072] = xb x Wcat. Block tile 128x128, K-step 64, double-buffered
// async-to-LDS staging; 8 waves, wave tile 64x32 (4x2 WMMA tiles).
__global__ __launch_bounds__(256) void perf_qkv(const bf16_t* __restrict__ xb,
                                                const bf16_t* __restrict__ wcat,
                                                bf16_t* __restrict__ qb,
                                                bf16_t* __restrict__ kb,
                                                bf16_t* __restrict__ vt) {
  __shared__ __align__(16) bf16_t xa_s[2][128 * 64];  // 2 x 16KB
  __shared__ __align__(16) bf16_t wb_s[2][128 * 64];  // 2 x 16KB

  const int M0 = blockIdx.y * 128;
  const int N0 = blockIdx.x * 128;
  const int tid = threadIdx.x;
  const int wave = tid >> 5;
  const int wm = wave >> 2;            // 0..1
  const int wn = wave & 3;             // 0..3
  const int rowb = 64 * wm;            // LDS-local row base
  const int colb = 32 * wn;            // LDS-local col base

  // Stage one K-step (128x64 of x and of w) into LDS buffer `buf`.
  // 8 async b128 instructions per wave per call (ASYNCcnt += 8).
  auto prefetch = [&](int buf, int k0) {
#pragma unroll
    for (int i = 0; i < 4; ++i) {
      const int c = tid + 256 * i;          // 0..1023 chunk id
      const int row = c >> 3;
      const int col8 = (c & 7) * 8;
      const bf16_t* gx = xb   + (size_t)(M0 + row) * DIM + k0 + col8;
      const bf16_t* gw = wcat + (size_t)(N0 + row) * DIM + k0 + col8;
      bf16_t* lx = &xa_s[buf][row * 64 + col8];
      bf16_t* lw = &wb_s[buf][row * 64 + col8];
#ifdef USE_ASYNC_LDS
      __builtin_amdgcn_global_load_async_to_lds_b128((as1_v4i*)gx, (as3_v4i*)lx, 0, 0);
      __builtin_amdgcn_global_load_async_to_lds_b128((as1_v4i*)gw, (as3_v4i*)lw, 0, 0);
#else
      *(v8bf*)lx = *(const v8bf*)gx;
      *(v8bf*)lw = *(const v8bf*)gw;
#endif
    }
  };

  v8f c[4][2];
#pragma unroll
  for (int i = 0; i < 4; ++i)
#pragma unroll
    for (int j = 0; j < 2; ++j) c[i][j] = vzero8();

  prefetch(0, 0);
  for (int ks = 0; ks < 16; ++ks) {
    // Always prefetch: ks==15 issues a dummy (reads land in adjacent ws
    // buffers, never consumed) so the in-loop wait can be an immediate.
    prefetch((ks + 1) & 1, (ks + 1) * 64);
#ifdef USE_ASYNC_LDS
    __builtin_amdgcn_s_wait_asynccnt(8);   // only the newest batch in flight
#endif
    __syncthreads();                        // current buffer visible to all

    const bf16_t* xl = &xa_s[ks & 1][0];
    const bf16_t* wl = &wb_s[ks & 1][0];
#pragma unroll
    for (int kk = 0; kk < 64; kk += 32) {
      v16bf a[4], b[2];
#pragma unroll
      for (int t = 0; t < 4; ++t) a[t] = load_a_frag(xl, 64, rowb + 16 * t, kk);
#pragma unroll
      for (int t = 0; t < 2; ++t) b[t] = load_b_frag(wl, 64, colb + 16 * t, kk);
#pragma unroll
      for (int i = 0; i < 4; ++i)
#pragma unroll
        for (int j = 0; j < 2; ++j) c[i][j] = wmma_bf16(a[i], b[j], c[i][j]);
    }
    __syncthreads();                        // done reading before overwrite
  }

  const int lane = tid & 31;
  const int cn = lane & 15, half = lane >> 4;
#pragma unroll
  for (int i = 0; i < 4; ++i) {
#pragma unroll
    for (int j = 0; j < 2; ++j) {
      const int col = N0 + colb + 16 * j + cn;   // 0..3071
      const int w = col >> 10;                   // 0=q,1=k,2=v
      const int cw = col & (DIM - 1);
      const int hh = cw >> 6, dd = cw & 63;
#pragma unroll
      for (int r = 0; r < 8; ++r) {
        const int rowg = M0 + rowb + 16 * i + r + 8 * half;  // 0..16383
        const int b_ = rowg >> 12;
        const int n_ = rowg & (NSEQ - 1);
        const int bh = b_ * HEADS + hh;
        const bf16_t v = (bf16_t)c[i][j][r];
        if (w == 0)      qb[((size_t)bh * NSEQ + n_) * HD + dd] = v;
        else if (w == 1) kb[((size_t)bh * NSEQ + n_) * HD + dd] = v;
        else             vt[((size_t)bh * HD + dd) * NSEQ + n_] = v;
      }
    }
  }
}

// ------------------------- fused phi(k) + kv ------------------------------
// Grid (bh, 8 n-splits), 8 waves. Loop over 512-row n-range in chunks of 32:
//   phase A: pk chunk via WMMA (K=64 over omega), exp, -> LDS pk^T [256,32]
//   phase B: kv[m,d] += pk^T_LDS x v^T  (single K=32 WMMA step)
// Writes fp32 partials kv_part[bh,ns,256,64] and s_part[bh,ns,256].
__global__ __launch_bounds__(256) void perf_kv(const bf16_t* __restrict__ kb,
                                               const bf16_t* __restrict__ vt,
                                               const bf16_t* __restrict__ omegab,
                                               const float* __restrict__ norm_k,
                                               float* __restrict__ kv_part,
                                               float* __restrict__ s_part) {
  const int bh = blockIdx.x;
  const int ns = blockIdx.y;
  const int nbeg = ns * (NSEQ / NSPLIT);
  const int nend = nbeg + (NSEQ / NSPLIT);

  __shared__ __align__(32) bf16_t pkT_s[MFEAT * 32];  // [m][n_local], 16KB
  __shared__ float norm_s[32];

  const int wave = threadIdx.x >> 5;
  const int lane = threadIdx.x & 31;
  const int cn = lane & 15, half = lane >> 4;
  const bf16_t* kbh = kb + (size_t)bh * NSEQ * HD;
  const bf16_t* vth = vt + (size_t)bh * HD * NSEQ;

  float s_acc[2] = {0.f, 0.f};
  v8f acc[2][4];
#pragma unroll
  for (int i = 0; i < 2; ++i)
#pragma unroll
    for (int j = 0; j < 4; ++j) acc[i][j] = vzero8();

  for (int n0 = nbeg; n0 < nend; n0 += 32) {
    if (threadIdx.x < 32) norm_s[threadIdx.x] = norm_k[(size_t)bh * NSEQ + n0 + threadIdx.x];
    __syncthreads();  // prev phase B done with pkT_s; norm_s visible after

    // ---- phase A: pk[n0..n0+32, 32w..32w+32) ----
    v8f c2[2][2];
#pragma unroll
    for (int tn = 0; tn < 2; ++tn)
#pragma unroll
      for (int tm = 0; tm < 2; ++tm) c2[tn][tm] = vzero8();
#pragma unroll
    for (int ks = 0; ks < 2; ++ks) {
      v16bf a[2], b[2];
#pragma unroll
      for (int tn = 0; tn < 2; ++tn) a[tn] = load_a_frag(kbh, HD, n0 + 16 * tn, 32 * ks);
#pragma unroll
      for (int tm = 0; tm < 2; ++tm)
        b[tm] = load_b_frag(omegab, HD, 32 * wave + 16 * tm, 32 * ks);
#pragma unroll
      for (int tn = 0; tn < 2; ++tn)
#pragma unroll
        for (int tm = 0; tm < 2; ++tm) c2[tn][tm] = wmma_bf16(a[tn], b[tm], c2[tn][tm]);
    }
#pragma unroll
    for (int tn = 0; tn < 2; ++tn) {
#pragma unroll
      for (int tm = 0; tm < 2; ++tm) {
        const int m = 32 * wave + 16 * tm + cn;
#pragma unroll
        for (int r = 0; r < 8; ++r) {
          const int nl = 16 * tn + r + 8 * half;
          const float val = __expf(c2[tn][tm][r] - norm_s[nl]) * RSQRT_M;
          s_acc[tm] += val;
          pkT_s[m * 32 + nl] = (bf16_t)val;
        }
      }
    }
    __syncthreads();

    // ---- phase B: kv[m,d] += pk^T x v^T ; K = 32 (one step) ----
    v16bf a[2], b[4];
#pragma unroll
    for (int tm = 0; tm < 2; ++tm) a[tm] = load_a_frag(pkT_s, 32, 32 * wave + 16 * tm, 0);
#pragma unroll
    for (int tn = 0; tn < 4; ++tn) b[tn] = load_b_frag(vth, NSEQ, 16 * tn, n0);
#pragma unroll
    for (int tm = 0; tm < 2; ++tm)
#pragma unroll
      for (int tn = 0; tn < 4; ++tn) acc[tm][tn] = wmma_bf16(a[tm], b[tn], acc[tm][tn]);
  }

  // ---- write fp32 partials ----
  float* kvp = kv_part + ((size_t)bh * NSPLIT + ns) * MFEAT * HD;
  float* sp  = s_part  + ((size_t)bh * NSPLIT + ns) * MFEAT;
#pragma unroll
  for (int tm = 0; tm < 2; ++tm) {
    float tot = s_acc[tm] + __shfl_xor(s_acc[tm], 16, 32);
    if (half == 0) sp[32 * wave + 16 * tm + cn] = tot;
  }
#pragma unroll
  for (int tm = 0; tm < 2; ++tm) {
#pragma unroll
    for (int tn = 0; tn < 4; ++tn) {
      const int d = 16 * tn + cn;
#pragma unroll
      for (int r = 0; r < 8; ++r) {
        const int m = 32 * wave + 16 * tm + r + 8 * half;
        kvp[m * HD + d] = acc[tm][tn][r];
      }
    }
  }
}

// Fold NSPLIT partials: kv^T [bh,64,256] bf16 and s [bh,256] fp32.
__global__ void perf_kv_reduce(const float* __restrict__ kv_part,
                               const float* __restrict__ s_part,
                               bf16_t* __restrict__ kvT,
                               float* __restrict__ sbuf) {
  int i = blockIdx.x * blockDim.x + threadIdx.x;
  if (i < BH * MFEAT * HD) {
    const int bh = i / (MFEAT * HD);
    const int r = i % (MFEAT * HD);
    const int m = r / HD, d = r % HD;
    float s = 0.f;
#pragma unroll
    for (int ns = 0; ns < NSPLIT; ++ns)
      s += kv_part[(((size_t)bh * NSPLIT + ns) * MFEAT + m) * HD + d];
    kvT[((size_t)bh * HD + d) * MFEAT + m] = (bf16_t)s;
  }
  if (i < BH * MFEAT) {
    const int bh = i / MFEAT, m = i % MFEAT;
    float s = 0.f;
#pragma unroll
    for (int ns = 0; ns < NSPLIT; ++ns)
      s += s_part[((size_t)bh * NSPLIT + ns) * MFEAT + m];
    sbuf[i] = s;
  }
}

// ------------------------ fused phi(q) + out GEMM -------------------------
// Grid (64 bh, 64 n-chunks of 64 rows); 4 waves, each owns 16 rows.
__global__ __launch_bounds__(128) void perf_out(const bf16_t* __restrict__ qb,
                                                const bf16_t* __restrict__ omegab,
                                                const bf16_t* __restrict__ kvT,
                                                const float* __restrict__ norm_q,
                                                const float* __restrict__ s_in,
                                                float* __restrict__ out) {
  const int bh = blockIdx.x;
  const int b = bh >> 4, h = bh & 15;
  const int wave = threadIdx.x >> 5;
  const int lane = threadIdx.x & 31;
  const int cn = lane & 15, half = lane >> 4;
  const int nw = blockIdx.y * 64 + 16 * wave;  // this wave's first row

  __shared__ __align__(32) bf16_t pq_s[4][16 * MFEAT];  // 4 x 8KB (wave-private)
  __shared__ float z_s[4][16];

  const bf16_t* qh  = qb  + (size_t)bh * NSEQ * HD;
  const bf16_t* kvh = kvT + (size_t)bh * HD * MFEAT;

  // per-lane norms for the 8 rows this lane touches in C tiles
  float nrm_l = norm_q[(size_t)bh * NSEQ + nw + (lane & 15)];
  float nrm[8];
#pragma unroll
  for (int r = 0; r < 8; ++r) nrm[r] = __shfl(nrm_l, r + 8 * half, 32);

  // ---- phase 1: pq tile [16, 256] = exp(q x omega^T - norm)/sqrt(M) ----
  v16bf a0 = load_a_frag(qh, HD, nw, 0);
  v16bf a1 = load_a_frag(qh, HD, nw, 32);
#pragma unroll
  for (int j = 0; j < 16; ++j) {
    v8f c = vzero8();
    c = wmma_bf16(a0, load_b_frag(omegab, HD, 16 * j, 0), c);
    c = wmma_bf16(a1, load_b_frag(omegab, HD, 16 * j, 32), c);
#pragma unroll
    for (int r = 0; r < 8; ++r) {
      const int rl = r + 8 * half;
      const float val = __expf(c[r] - nrm[r]) * RSQRT_M;
      pq_s[wave][rl * MFEAT + 16 * j + cn] = (bf16_t)val;
    }
  }

  // ---- z = pq . s (from LDS; lane pair covers 128 m each) ----
  {
    float zp = 0.f;
    const float* sv = s_in + (size_t)bh * MFEAT + 128 * half;
    const bf16_t* pr = &pq_s[wave][cn * MFEAT + 128 * half];
    for (int i = 0; i < 128; ++i) zp += (float)pr[i] * sv[i];
    zp += __shfl_xor(zp, 16, 32);
    if (half == 0) z_s[wave][cn] = zp;
  }

  // ---- phase 2: out[16, 64] = pq x kv ; K = 256 ----
  v8f acc[4];
#pragma unroll
  for (int tn = 0; tn < 4; ++tn) acc[tn] = vzero8();
#pragma unroll
  for (int ks = 0; ks < 8; ++ks) {
    v16bf a = load_a_frag(pq_s[wave], MFEAT, 0, 32 * ks);
#pragma unroll
    for (int tn = 0; tn < 4; ++tn) {
      v16bf bfrag = load_b_frag(kvh, MFEAT, 16 * tn, 32 * ks);
      acc[tn] = wmma_bf16(a, bfrag, acc[tn]);
    }
  }

  // ---- epilogue: divide by z, store fp32 [b, n, 1024] ----
#pragma unroll
  for (int tn = 0; tn < 4; ++tn) {
    const int d = 16 * tn + cn;
#pragma unroll
    for (int r = 0; r < 8; ++r) {
      const int rl = r + 8 * half;
      const float z = z_s[wave][rl];
      out[((size_t)b * NSEQ + nw + rl) * DIM + h * HD + d] = acc[tn][r] / (z + 1e-6f);
    }
  }
}

// ------------------------------- launcher ---------------------------------

extern "C" void kernel_launch(void* const* d_in, const int* in_sizes, int n_in,
                              void* d_out, int out_size, void* d_ws, size_t ws_size,
                              hipStream_t stream) {
  (void)in_sizes; (void)n_in; (void)out_size; (void)ws_size;
  const float* x     = (const float*)d_in[0];
  const float* Wq    = (const float*)d_in[1];
  const float* Wk    = (const float*)d_in[2];
  const float* Wv    = (const float*)d_in[3];
  const float* omega = (const float*)d_in[4];
  float* out = (float*)d_out;
  char* ws = (char*)d_ws;

  // workspace carve-up (all offsets 256B-aligned); total ~178 MB
  size_t off = 0;
  bf16_t* xb      = (bf16_t*)(ws + off); off += (size_t)ROWS * DIM * 2;          // 32 MB
  bf16_t* wcat    = (bf16_t*)(ws + off); off += (size_t)3 * DIM * DIM * 2;       // 6 MB
  bf16_t* omegab  = (bf16_t*)(ws + off); off += (size_t)MFEAT * HD * 2;          // 32 KB
  bf16_t* qb      = (bf16_t*)(ws + off); off += (size_t)BH * NSEQ * HD * 2;      // 32 MB
  bf16_t* kb      = (bf16_t*)(ws + off); off += (size_t)BH * NSEQ * HD * 2;      // 32 MB
  bf16_t* vt      = (bf16_t*)(ws + off); off += (size_t)BH * HD * NSEQ * 2;      // 32 MB
  float*  norm_q  = (float*)(ws + off);  off += (size_t)BH * NSEQ * 4;           // 1 MB
  float*  norm_k  = (float*)(ws + off);  off += (size_t)BH * NSEQ * 4;           // 1 MB
  bf16_t* kvT     = (bf16_t*)(ws + off); off += (size_t)BH * HD * MFEAT * 2;     // 2 MB
  float*  sbuf    = (float*)(ws + off);  off += (size_t)BH * MFEAT * 4;          // 64 KB
  float*  kv_part = (float*)(ws + off);  off += (size_t)BH * NSPLIT * MFEAT * HD * 4; // 32 MB
  float*  s_part  = (float*)(ws + off);  off += (size_t)BH * NSPLIT * MFEAT * 4; // 512 KB

  const int nx = ROWS * DIM;
  perf_cvt_x<<<(nx + 255) / 256, 256, 0, stream>>>(x, xb, nx);
  perf_cvt_wt<<<(3 * DIM * DIM + 255) / 256, 256, 0, stream>>>(Wq, Wk, Wv, wcat);
  perf_cvt_omega<<<(MFEAT * HD + 255) / 256, 256, 0, stream>>>(omega, omegab, MFEAT * HD);

  perf_qkv<<<dim3(3 * DIM / 128, ROWS / 128), 256, 0, stream>>>(xb, wcat, qb, kb, vt);

  perf_norms<<<(BH * NSEQ + 255) / 256, 256, 0, stream>>>(qb, norm_q, BH * NSEQ);
  perf_norms<<<(BH * NSEQ + 255) / 256, 256, 0, stream>>>(kb, norm_k, BH * NSEQ);

  perf_kv<<<dim3(BH, NSPLIT), 256, 0, stream>>>(kb, vt, omegab, norm_k, kv_part, s_part);
  perf_kv_reduce<<<(BH * MFEAT * HD + 255) / 256, 256, 0, stream>>>(kv_part, s_part, kvT, sbuf);

  perf_out<<<dim3(BH, NSEQ / 64), 128, 0, stream>>>(qb, omegab, kvT, norm_q, sbuf, out);
}
